// ShepardsLayer_90890097918518
// MI455X (gfx1250) — compile-verified, matrix operand order checked by reading, not run on hardware
//
#include <hip/hip_runtime.h>

// ---------------------------------------------------------------------------
// Shepard's-attention layer for MI455X (gfx1250, wave32, WMMA + TDM).
// All contractions run on v_wmma_f32_16x16x32_bf16; fp32 row norms carry the
// precision-critical part of the distance; V is produced directly in a
// key-blocked transposed layout so attention tiles are contiguous; attention
// K/V tiles are staged by the Tensor Data Mover into double-buffered LDS
// (TENSORcnt-tracked, overlapped with WMMA compute).
// ---------------------------------------------------------------------------

typedef __attribute__((ext_vector_type(16))) __bf16        v16bf;
typedef __attribute__((ext_vector_type(8)))  float         v8f;
typedef __attribute__((ext_vector_type(4)))  unsigned int  u32x4;
typedef __attribute__((ext_vector_type(8)))  int           i32x8;
typedef __attribute__((ext_vector_type(4)))  int           i32x4;

#define DEV __device__ __forceinline__

constexpr int Bb   = 4;
constexpr int Ss   = 2048;
constexpr int Dd   = 1024;
constexpr int DQKc = 128;
constexpr int DVc  = 256;
constexpr int Mtot = Bb * Ss;           // 8192 rows total
constexpr float EPS2 = 1e-16f;          // eps^2 : W = 1/max(d2, eps^2)

// ---- WMMA fragment helpers -------------------------------------------------
// A-fragment (16xK rows in lanes) and B-fragment from [N][K]-transposed
// storage share the same per-lane layout:
//   lane L: row = L & 15, holds K = kb..kb+7 and kb+16..kb+23, kb = (L>=16)*8
// With a row-major [row][ld] source this is two contiguous 16B ds_load_b128s.
DEV v16bf load_frag(const __bf16* p, int ld) {
  const int lane = threadIdx.x & 31;
  const int row  = lane & 15;
  const int kb   = (lane >> 4) << 3;    // 0 or 8
  const __bf16* q = p + row * ld + kb;
  union { v16bf v; float4 f[2]; } u;
  u.f[0] = *(const float4*)(q);
  u.f[1] = *(const float4*)(q + 16);
  return u.v;
}

DEV v8f wmma_bf16(v16bf a, v16bf b, v8f c) {
  // D = A(16x32) * B(32x16) + C, f32 accumulate
  return __builtin_amdgcn_wmma_f32_16x16x32_bf16(
      false, a, false, b, (short)0, c, false, false);
}

// ---- Tensor Data Mover: 2-D tile (bf16) global -> LDS ----------------------
// D# packed per CDNA5 ISA 8.3/8.4: group0 = {count=1, lds_addr, global_addr,
// type=2}; group1 = {data_size=2B, tensor_dim0/1, tile_dim0/1, dim0_stride}.
// Groups 2/3 zero (2-D tensor). Tracked by TENSORcnt.
// This toolchain's builtin takes 6 args (g0, g1, g2, g3, g_extra, cpol).
DEV void tdm_load_2d(unsigned lds_off, const void* gaddr,
                     unsigned tensor_d0, unsigned tensor_d1,
                     unsigned tile_d0, unsigned tile_d1, unsigned stride0) {
  u32x4 g0;
  g0[0] = 1u;                                   // count=1 (valid descriptor)
  g0[1] = lds_off;                              // lds_addr (bytes)
  unsigned long long ga = (unsigned long long)gaddr;
  g0[2] = (unsigned)(ga & 0xffffffffu);         // global_addr[31:0]
  g0[3] = (unsigned)((ga >> 32) & 0x1ffffffu)   // global_addr[56:32]
          | (2u << 30);                         // type=2 ("image")
  i32x8 g1;
  g1[0] = (int)(1u << 16);                      // data_size=1 -> 2 bytes
  g1[1] = (int)((tensor_d0 & 0xffffu) << 16);   // tensor_dim0[15:0]
  g1[2] = (int)((tensor_d0 >> 16) | ((tensor_d1 & 0xffffu) << 16));
  g1[3] = (int)((tensor_d1 >> 16) | (tile_d0 << 16));   // tile_dim0
  g1[4] = (int)(tile_d1 & 0xffffu);             // tile_dim1 (tile_dim2 = 0)
  g1[5] = (int)stride0;                         // tensor_dim0_stride[31:0]
  g1[6] = 0;                                    // stride0 hi / stride1 lo
  g1[7] = 0;
  i32x4 z4 = {0, 0, 0, 0};
  i32x8 z8 = {0, 0, 0, 0, 0, 0, 0, 0};
  __builtin_amdgcn_tensor_load_to_lds(g0, g1, z4, z4, z8, 0);
}

// ---- prep kernels ----------------------------------------------------------
__global__ __launch_bounds__(256) void cvt_bf16_kernel(
    const float* __restrict__ in, __bf16* __restrict__ out, int n) {
  int i = blockIdx.x * 256 + threadIdx.x;
  if (i < n) out[i] = (__bf16)in[i];
}

// W[K][N] (row-major) -> Wt[N][K] bf16 so GEMM B-tiles load contiguously.
__global__ __launch_bounds__(256) void transpose_cvt_kernel(
    const float* __restrict__ W, __bf16* __restrict__ Wt, int K, int N) {
  int i = blockIdx.x * 256 + threadIdx.x;
  if (i < K * N) {
    int k = i / N, n = i - k * N;
    Wt[(size_t)n * K + k] = (__bf16)W[i];
  }
}

// ---- generic bf16 GEMM: C[M,N] = A[M,Kd] * Bt[N,Kd]^T + bias --------------
// Block: 256 threads (8 waves); tile 128(M) x 128(N); K-step 32.
// Wave w computes rows [w*16, w*16+16) x all 128 columns (8 WMMA accums).
// STORE_VBLK writes bf16 output in key-blocked transposed layout
// Vt[b][s/32][n][s%32] so the attention kernel can stage V tiles with a
// single contiguous TDM transfer.
template<bool STORE_F32, bool STORE_BF16, bool SUMSQ, bool ADDX, bool STORE_VBLK>
__global__ __launch_bounds__(256) void gemm_kernel(
    const __bf16* __restrict__ A,    // [M][Kd]
    const __bf16* __restrict__ Bt,   // [N][Kd] (pre-transposed)
    const float*  __restrict__ bias, // [N]
    const float*  __restrict__ Xres, // [M][N] residual (ADDX) or null
    float*  __restrict__ outF,
    __bf16* __restrict__ outB,
    float*  __restrict__ ssq,        // [M] row sum-of-squares (SUMSQ)
    int Kd, int N) {
  __shared__ __bf16 As[128 * 32];
  __shared__ __bf16 Bs[128 * 32];

  const int tid  = threadIdx.x;
  const int wave = tid >> 5;
  const int lane = tid & 31;
  const int hi   = (lane >> 4) & 1;
  const int mbase = blockIdx.y * 128;
  const int nbase = blockIdx.x * 128;

  v8f acc[8];
  v8f zero = {};
#pragma unroll
  for (int t = 0; t < 8; ++t) acc[t] = zero;

  const int lrow = tid >> 1;           // 0..127 : tile row loaded by this thread
  const int lseg = (tid & 1) << 4;     // 0 or 16 elements (32B halves)

  for (int kb = 0; kb < Kd; kb += 32) {
    __syncthreads();
    {
      const __bf16* src = A + (size_t)(mbase + lrow) * Kd + kb + lseg;
      float4* dst = (float4*)(As + lrow * 32 + lseg);
      dst[0] = ((const float4*)src)[0];
      dst[1] = ((const float4*)src)[1];
    }
    {
      const __bf16* src = Bt + (size_t)(nbase + lrow) * Kd + kb + lseg;
      float4* dst = (float4*)(Bs + lrow * 32 + lseg);
      dst[0] = ((const float4*)src)[0];
      dst[1] = ((const float4*)src)[1];
    }
    __syncthreads();

    // Prefetch next K-step tiles into cache while WMMAs run.
    if (kb + 32 < Kd) {
      __builtin_prefetch(A + (size_t)(mbase + lrow) * Kd + kb + 32 + lseg, 0, 0);
      __builtin_prefetch(Bt + (size_t)(nbase + lrow) * Kd + kb + 32 + lseg, 0, 0);
    }

    v16bf a = load_frag(As + wave * 16 * 32, 32);
#pragma unroll
    for (int t = 0; t < 8; ++t) {
      v16bf b = load_frag(Bs + t * 16 * 32, 32);
      acc[t] = wmma_bf16(a, b, acc[t]);
    }
  }

  // Epilogue. C layout: lane holds n = (lane&15)+16t, m = wave*16 + hi*8 + r.
  float ssqp[8];
  if (SUMSQ) {
#pragma unroll
    for (int r = 0; r < 8; ++r) ssqp[r] = 0.f;
  }
#pragma unroll
  for (int t = 0; t < 8; ++t) {
    const int n_g = nbase + t * 16 + (lane & 15);
    const float bv = bias[n_g];
#pragma unroll
    for (int r = 0; r < 8; ++r) {
      const size_t m_g = (size_t)(mbase + wave * 16 + hi * 8 + r);
      float v = acc[t][r] + bv;
      if (ADDX)       v += Xres[m_g * N + n_g];
      if (STORE_F32)  outF[m_g * N + n_g] = v;
      if (STORE_BF16) outB[m_g * N + n_g] = (__bf16)v;
      if (STORE_VBLK) {
        const int bidx = (int)(m_g >> 11);        // / Ss
        const int srow = (int)(m_g & 2047);       // % Ss
        const size_t d =
            (((size_t)bidx * (Ss / 32) + (srow >> 5)) * DVc + n_g) * 32 +
            (srow & 31);
        outB[d] = (__bf16)v;
      }
      if (SUMSQ)      ssqp[r] += v * v;
    }
  }
  if (SUMSQ) {  // valid when one n-tile covers the whole row (N == 128)
#pragma unroll
    for (int r = 0; r < 8; ++r) {
      float s = ssqp[r];
      s += __shfl_xor(s, 1);
      s += __shfl_xor(s, 2);
      s += __shfl_xor(s, 4);
      s += __shfl_xor(s, 8);
      if ((lane & 15) == 0)
        ssq[mbase + wave * 16 + hi * 8 + r] = s;
    }
  }
}

// ---- Shepard attention core ------------------------------------------------
// Grid: (S/128 q-tiles, B). Block: 256 threads = 8 waves; wave owns 16 q-rows.
// Keys stream in chunks of 32 through double-buffered LDS filled by the TDM
// (wave 0 issues next chunk's descriptors, all waves compute current chunk):
// s = Q.K^T via WMMA, w = rcp(max(d2,eps^2)) bf16-rounded, w restaged through
// a per-wave LDS tile into A-fragment layout, acc += w * V via WMMA.
// Output fused with gate G, stored bf16 for the output projection.
__global__ __launch_bounds__(256) void attn_kernel(
    const __bf16* __restrict__ Qb,  // [M][128] bf16
    const __bf16* __restrict__ Kb,  // [M][128] bf16
    const __bf16* __restrict__ Vt,  // [B][S/32][256][32] bf16 (key-blocked ^T)
    const float*  __restrict__ q2,  // [M] ||q||^2 (incl. bias)
    const float*  __restrict__ k2,  // [M] ||k||^2
    const float*  __restrict__ G,   // [M][256] f32 gate
    __bf16* __restrict__ AGb)       // [M][256] bf16  (A * G)
{
  __shared__ __bf16 ldsK[2][32 * DQKc];    // [buf][key][dqk]   2 x  8 KB
  __shared__ __bf16 ldsVt[2][DVc * 32];    // [buf][n][key]     2 x 16 KB
  __shared__ __bf16 wbuf[8 * 16 * 32];     // per-wave w tile        8 KB

  const int tid  = threadIdx.x;
  const int wave = tid >> 5;
  const int lane = tid & 31;
  const int hi   = (lane >> 4) & 1;
  const int b    = blockIdx.y;
  const int q0   = blockIdx.x * 128 + wave * 16;   // query row in batch
  const size_t mg0 = (size_t)b * Ss + q0;

  // Preload this wave's Q fragments (4 K-chunks of 32) straight from global.
  v16bf qf[4];
#pragma unroll
  for (int c = 0; c < 4; ++c)
    qf[c] = load_frag(Qb + mg0 * DQKc + c * 32, DQKc);

  float q2r[8];
#pragma unroll
  for (int r = 0; r < 8; ++r) q2r[r] = q2[mg0 + hi * 8 + r];

  v8f accv[16];
  v8f zero = {};
#pragma unroll
  for (int t = 0; t < 16; ++t) accv[t] = zero;
  float rsum[8] = {};

  constexpr int nch = Ss / 32;  // 64 key chunks
  // TDM issue for chunk i into buffer bf (two instructions per call: K tile + V tile, wave 0).
  auto issue = [&](int i, int bf) {
    const __bf16* gK = Kb + ((size_t)b * Ss + (size_t)i * 32) * DQKc;
    const __bf16* gV = Vt + ((size_t)b * nch + i) * ((size_t)DVc * 32);
    tdm_load_2d((unsigned)(size_t)&ldsK[bf][0], gK,
                DQKc, 32, DQKc, 32, DQKc);               // 32x128 2-D tile
    tdm_load_2d((unsigned)(size_t)&ldsVt[bf][0], gV,
                DVc * 32, 1, DVc * 32, 1, DVc * 32);     // contiguous 16 KB
  };

  if (tid < 32) {                    // wave 0 primes buffer 0
    issue(0, 0);
    __builtin_amdgcn_s_wait_tensorcnt(0);
  }
  __syncthreads();

  for (int i = 0; i < nch; ++i) {
    const int bf = i & 1;
    if (tid < 32 && (i + 1) < nch) issue(i + 1, bf ^ 1);  // overlap next DMA

    // s = Q . K^T for the two 16-key halves, then Shepard weights.
#pragma unroll
    for (int nt = 0; nt < 2; ++nt) {
      v8f s = zero;
#pragma unroll
      for (int c = 0; c < 4; ++c) {
        v16bf kf = load_frag(&ldsK[bf][nt * 16 * DQKc + c * 32], DQKc);
        s = wmma_bf16(qf[c], kf, s);
      }
      const float k2n =
          k2[(size_t)b * Ss + (size_t)i * 32 + nt * 16 + (lane & 15)];
#pragma unroll
      for (int r = 0; r < 8; ++r) {
        const float d2 = q2r[r] + k2n - 2.0f * s[r];
        // w = max(dist,eps)^-2 = 1/max(d2,eps^2); v_rcp_f32 is exact enough
        // since w is consumed at bf16 precision by the WMMA.
        const float w = __builtin_amdgcn_rcpf(fmaxf(d2, EPS2));
        const __bf16 wb = (__bf16)w;
        rsum[r] += (float)wb;   // normalizer matches what the matmul sums
        wbuf[wave * 512 + (hi * 8 + r) * 32 + nt * 16 + (lane & 15)] = wb;
      }
    }
    // Per-wave LDS round-trip (C-layout -> A-fragment); DS ops are in-order
    // per wave, the explicit wait makes the RAW airtight.
    asm volatile("s_wait_dscnt 0" ::: "memory");
    v16bf wf = load_frag(wbuf + wave * 512, 32);
#pragma unroll
    for (int t = 0; t < 16; ++t) {
      v16bf vf = load_frag(&ldsVt[bf][t * 16 * 32], 32);
      accv[t] = wmma_bf16(wf, vf, accv[t]);
    }

    if (tid < 32) __builtin_amdgcn_s_wait_tensorcnt(0);  // next buffer landed
    __syncthreads();   // publish buffer bf^1; all waves done reading bf
  }

  // Normalize rows, fuse gate G, store bf16 A*G.
#pragma unroll
  for (int r = 0; r < 8; ++r) {
    float s = rsum[r];
    s += __shfl_xor(s, 1);
    s += __shfl_xor(s, 2);
    s += __shfl_xor(s, 4);
    s += __shfl_xor(s, 8);
    rsum[r] = __builtin_amdgcn_rcpf(s);
  }
#pragma unroll
  for (int t = 0; t < 16; ++t) {
    const int n = t * 16 + (lane & 15);
#pragma unroll
    for (int r = 0; r < 8; ++r) {
      const size_t mg = mg0 + hi * 8 + r;
      const float av = accv[t][r] * rsum[r];
      AGb[mg * DVc + n] = (__bf16)(av * G[mg * DVc + n]);
    }
  }
}

// ---------------------------------------------------------------------------
extern "C" void kernel_launch(void* const* d_in, const int* in_sizes, int n_in,
                              void* d_out, int out_size, void* d_ws,
                              size_t ws_size, hipStream_t stream) {
  const float* X  = (const float*)d_in[0];
  const float* Wq = (const float*)d_in[1];
  const float* bq = (const float*)d_in[2];
  const float* Wk = (const float*)d_in[3];
  const float* bk = (const float*)d_in[4];
  const float* Wv = (const float*)d_in[5];
  const float* bv = (const float*)d_in[6];
  const float* Wg = (const float*)d_in[7];
  const float* bg = (const float*)d_in[8];
  const float* Wp = (const float*)d_in[9];
  const float* bp = (const float*)d_in[10];
  float* out = (float*)d_out;

  // Workspace carve-up (~38 MB total, fully L2-resident on MI455X).
  char* ws = (char*)d_ws;
  auto alloc = [&](size_t bytes) -> void* {
    void* p = ws;
    ws += (bytes + 255) & ~(size_t)255;
    return p;
  };
  __bf16* Xb  = (__bf16*)alloc((size_t)Mtot * Dd * 2);
  __bf16* Wqt = (__bf16*)alloc((size_t)Dd * DQKc * 2);
  __bf16* Wkt = (__bf16*)alloc((size_t)Dd * DQKc * 2);
  __bf16* Wvt = (__bf16*)alloc((size_t)Dd * DVc * 2);
  __bf16* Wgt = (__bf16*)alloc((size_t)Dd * DVc * 2);
  __bf16* Wpt = (__bf16*)alloc((size_t)DVc * Dd * 2);
  __bf16* Qb  = (__bf16*)alloc((size_t)Mtot * DQKc * 2);
  __bf16* Kbf = (__bf16*)alloc((size_t)Mtot * DQKc * 2);
  __bf16* Vtb = (__bf16*)alloc((size_t)Mtot * DVc * 2);   // blocked ^T layout
  float*  Gf  = (float*) alloc((size_t)Mtot * DVc * 4);
  float*  q2  = (float*) alloc((size_t)Mtot * 4);
  float*  k2  = (float*) alloc((size_t)Mtot * 4);
  __bf16* AGb = (__bf16*)alloc((size_t)Mtot * DVc * 2);

  // 1) bf16 conversions / weight transposes (one-time, tiny vs. GEMMs)
  cvt_bf16_kernel<<<(Mtot * Dd + 255) / 256, 256, 0, stream>>>(X, Xb, Mtot * Dd);
  transpose_cvt_kernel<<<(Dd * DQKc + 255) / 256, 256, 0, stream>>>(Wq, Wqt, Dd, DQKc);
  transpose_cvt_kernel<<<(Dd * DQKc + 255) / 256, 256, 0, stream>>>(Wk, Wkt, Dd, DQKc);
  transpose_cvt_kernel<<<(Dd * DVc  + 255) / 256, 256, 0, stream>>>(Wv, Wvt, Dd, DVc);
  transpose_cvt_kernel<<<(Dd * DVc  + 255) / 256, 256, 0, stream>>>(Wg, Wgt, Dd, DVc);
  transpose_cvt_kernel<<<(DVc * Dd  + 255) / 256, 256, 0, stream>>>(Wp, Wpt, DVc, Dd);

  // 2) projections (WMMA GEMMs); Q/K also emit fp32 row norms for distances
  dim3 gq(DQKc / 128, Mtot / 128);
  gemm_kernel<false, true, true, false, false><<<gq, 256, 0, stream>>>(
      Xb, Wqt, bq, nullptr, nullptr, Qb, q2, Dd, DQKc);
  gemm_kernel<false, true, true, false, false><<<gq, 256, 0, stream>>>(
      Xb, Wkt, bk, nullptr, nullptr, Kbf, k2, Dd, DQKc);
  dim3 gv(DVc / 128, Mtot / 128);
  gemm_kernel<false, false, false, false, true><<<gv, 256, 0, stream>>>(
      Xb, Wvt, bv, nullptr, nullptr, Vtb, nullptr, Dd, DVc);
  gemm_kernel<true, false, false, false, false><<<gv, 256, 0, stream>>>(
      Xb, Wgt, bg, nullptr, Gf, nullptr, nullptr, Dd, DVc);

  // 3) streaming Shepard attention fused with the G gate (TDM-fed)
  dim3 ga(Ss / 128, Bb);
  attn_kernel<<<ga, 256, 0, stream>>>(Qb, Kbf, Vtb, q2, k2, Gf, AGb);

  // 4) output projection + residual: out = X + (A*G) @ Wp + bp
  dim3 go(Dd / 128, Mtot / 128);
  gemm_kernel<true, false, false, true, false><<<go, 256, 0, stream>>>(
      AGb, Wpt, bp, X, out, nullptr, nullptr, DVc, Dd);
}
